// FieldEMACE_80290118631833
// MI455X (gfx1250) — compile-verified, matrix operand order checked by reading, not run on hardware
//
#include <hip/hip_runtime.h>
#include <hip/hip_bf16.h>

// ---------------------------------------------------------------------------
// Sizes (from reference): N=10000 nodes, E=160000 edges, G=16 graphs,
// NEL=10, C=64, NSH=16, NB=8, NLAYERS=2, NSTATES=3, M_MM=1024.
// ---------------------------------------------------------------------------

typedef __attribute__((ext_vector_type(16))) _Float16 v16h;
typedef __attribute__((ext_vector_type(8)))  _Float16 v8h;
typedef __attribute__((ext_vector_type(8)))  float    v8f;
typedef __attribute__((ext_vector_type(2)))  float    v2f;
typedef __attribute__((ext_vector_type(4)))  unsigned int u32x4;
typedef __attribute__((ext_vector_type(8)))  int      i32x8;
typedef __attribute__((ext_vector_type(4)))  int      i32x4;

#define R_MAX 5.0f
#define AVG_NEIGH 16.0f

#if defined(__has_builtin)
#if __has_builtin(__builtin_amdgcn_tensor_load_to_lds) && __has_builtin(__builtin_amdgcn_s_wait_tensorcnt)
#define HAVE_TDM 1
#endif
#if __has_builtin(__builtin_amdgcn_wmma_f32_16x16x4_f32)
#define HAVE_WMMA_F32 1
#endif
#endif

__device__ __forceinline__ float fast_silu(float x) {
  return x * __builtin_amdgcn_rcpf(1.0f + __expf(-x));
}

__device__ __forceinline__ v8f wmma_f16(v16h a, v16h b, v8f c) {
  return __builtin_amdgcn_wmma_f32_16x16x32_f16(
      false, a, false, b, (short)0, c, false, false);
}

#if defined(HAVE_WMMA_F32)
__device__ __forceinline__ v8f wmma_f32(v2f a, v2f b, v8f c) {
  // D = A(16x4 f32) * B(4x16 f32) + C(16x16 f32)
  return __builtin_amdgcn_wmma_f32_16x16x4_f32(
      false, a, false, b, (short)0, c, false, false);
}
#endif

// A fragment (16x32 f16 chunk of row-major [16][ldk] LDS array, K offset k0).
__device__ __forceinline__ v16h load_frag_A(const _Float16* ld, int ldk, int k0, int lane) {
  int half = lane >> 4;
  int m = lane & 15;
  const _Float16* p = ld + m * ldk + k0 + half * 8;
  v8h lo = *(const v8h*)(p);
  v8h hi = *(const v8h*)(p + 16);
  return __builtin_shufflevector(lo, hi, 0, 1, 2, 3, 4, 5, 6, 7,
                                 8, 9, 10, 11, 12, 13, 14, 15);
}

// B fragment from fragment-major (pre-swizzled) LDS.
__device__ __forceinline__ v16h load_frag_Bsw(const _Float16* ld, int frag, int lane) {
  const _Float16* p = ld + (frag * 32 + lane) * 16;
  v8h lo = *(const v8h*)(p);
  v8h hi = *(const v8h*)(p + 8);
  return __builtin_shufflevector(lo, hi, 0, 1, 2, 3, 4, 5, 6, 7,
                                 8, 9, 10, 11, 12, 13, 14, 15);
}

// Stage one 8KB W3 chunk into LDS: TDM if available, else vector copy.
__device__ __forceinline__ void stage_w3_chunk(const _Float16* __restrict__ src,
                                               _Float16* dst, int tid) {
#if defined(HAVE_TDM)
  if (tid == 0) {
    unsigned lds_off = (unsigned)(size_t)dst;
    unsigned long long ga = (unsigned long long)(size_t)src;
    u32x4 g0;
    g0[0] = 1u;                                   // count=1
    g0[1] = lds_off;                              // lds_addr
    g0[2] = (unsigned)(ga & 0xffffffffu);         // global_addr[31:0]
    g0[3] = (unsigned)((ga >> 32) & 0x01ffffffu) | 0x80000000u; // type=2
    i32x8 g1;
    g1[0] = 1 << 16;            // data_size=1 (2 bytes)
    g1[1] = (int)(4096u << 16); // tensor_dim0 = 4096
    g1[2] = 0;
    g1[3] = 4096 << 16;         // tile_dim0 = 4096
    g1[4] = 0;
    g1[5] = 4096;               // tensor_dim0_stride
    g1[6] = 0;
    g1[7] = 0;
    i32x4 gz = {0, 0, 0, 0};
#if __clang_major__ >= 23
    i32x8 gz8 = {0, 0, 0, 0, 0, 0, 0, 0};
    __builtin_amdgcn_tensor_load_to_lds(g0, g1, gz, gz, gz8, 0);
#else
    __builtin_amdgcn_tensor_load_to_lds(g0, g1, gz, gz, 0);
#endif
  }
#else
  for (int i = tid; i < 512; i += 128)
    ((uint4*)dst)[i] = ((const uint4*)src)[i];
#endif
}

__device__ __forceinline__ void wait_w3() {
#if defined(HAVE_TDM)
  __builtin_amdgcn_s_wait_tensorcnt(0);
#endif
}

// real spherical harmonics up to l=3 (matches reference); returns r
__device__ __forceinline__ float sph16(float vx, float vy, float vz, float* o) {
  float r = sqrtf(vx * vx + vy * vy + vz * vz + 1e-12f);
  float x = vx / r, y = vy / r, z = vz / r;
  const float s3 = 1.7320508075688772f, s5 = 2.23606797749979f, s7 = 2.6457513110645907f;
  const float s15 = 3.872983346207417f, s105 = 10.246950765959598f;
  const float c358 = 2.091650066335189f, c218 = 1.6201851746019651f;
  o[0] = 1.0f;
  o[1] = s3 * x; o[2] = s3 * y; o[3] = s3 * z;
  o[4] = s15 * x * y; o[5] = s15 * y * z; o[6] = 0.5f * s5 * (3.f * z * z - 1.f);
  o[7] = s15 * x * z; o[8] = 0.5f * s15 * (x * x - y * y);
  o[9] = c358 * y * (3.f * x * x - y * y);
  o[10] = s105 * x * y * z;
  o[11] = c218 * y * (5.f * z * z - 1.f);
  o[12] = 0.5f * s7 * z * (5.f * z * z - 3.f);
  o[13] = c218 * x * (5.f * z * z - 1.f);
  o[14] = 0.5f * s105 * z * (x * x - y * y);
  o[15] = c358 * x * (x * x - 3.f * y * y);
  return r;
}

// ---------------------------------------------------------------------------
__global__ void zero_kernel(float* __restrict__ p, size_t n) {
  size_t i = (size_t)blockIdx.x * blockDim.x + threadIdx.x;
  if (i < n) p[i] = 0.0f;
}

// W2 (2,64,64) f32 -> f16 fragment-major per layer.
__global__ void swz_W2_kernel(const float* __restrict__ src, _Float16* __restrict__ dst) {
  int i = blockIdx.x * 256 + threadIdx.x;
  if (i >= 2 * 4096) return;
  int layer = i >> 12, r = i & 4095;
  int k = r >> 6, n = r & 63;
  int kc = k >> 5, kk = k & 31;
  int half = (kk >> 3) & 1, h = ((kk >> 4) & 1) * 8 + (kk & 7);
  int t = n >> 4, lane = half * 16 + (n & 15);
  int frag = t * 2 + kc;
  dst[layer * 4096 + (frag * 32 + lane) * 16 + h] = (_Float16)src[i];
}

// W3 (2,64,1024) f32 -> f16 chunked fragment-major.
__global__ void swz_W3_kernel(const float* __restrict__ src, _Float16* __restrict__ dst) {
  int i = blockIdx.x * 256 + threadIdx.x;
  if (i >= 2 * 65536) return;
  int layer = i >> 16, r = i & 65535;
  int k = r >> 10, n = r & 1023;
  int chunk = n >> 6, nn = n & 63;
  int kc = k >> 5, kk = k & 31;
  int half = (kk >> 3) & 1, h = ((kk >> 4) & 1) * 8 + (kk & 7);
  int t = nn >> 4, lane = half * 16 + (nn & 15);
  int frag = t * 2 + kc;
  dst[((size_t)(layer * 16 + chunk) * 8 + frag) * 512 + lane * 16 + h] = (_Float16)src[i];
}

// single block: amp = sum over 9 moments over all MM particles
__global__ void mm_moments_kernel(const float* __restrict__ mm_pos,
                                  const float* __restrict__ mm_q,
                                  float* __restrict__ amp_ws, int M) {
  __shared__ float red[256];
  float local = 0.0f;
  for (int i = threadIdx.x; i < M; i += 256) {
    float o[16];
    float r = sph16(mm_pos[3 * i], mm_pos[3 * i + 1], mm_pos[3 * i + 2], o);
    float q = mm_q[i];
    float r2 = r * r;
    float rl[9] = {1.f, r, r, r, r2, r2, r2, r2, r2};
#pragma unroll
    for (int m = 0; m < 9; ++m) local += q * rl[m] * o[m];
  }
  red[threadIdx.x] = local;
  __syncthreads();
  for (int s = 128; s > 0; s >>= 1) {
    if (threadIdx.x < s) red[threadIdx.x] += red[threadIdx.x + s];
    __syncthreads();
  }
  if (threadIdx.x == 0) amp_ws[0] = red[0];
}

__global__ void node_init_kernel(const float* __restrict__ pos,
                                 const float* __restrict__ attrs,
                                 const float* __restrict__ emb_W,
                                 const float* __restrict__ atomic_E,
                                 const float* __restrict__ field_up_W,
                                 const float* __restrict__ amp_ws,
                                 const int* __restrict__ batch,
                                 float* __restrict__ node_feats,
                                 float* __restrict__ mpf,
                                 float* __restrict__ e0_ws, int n) {
  int i = blockIdx.x * blockDim.x + threadIdx.x;
  if (i >= n) return;
  float a[10];
#pragma unroll
  for (int k = 0; k < 10; ++k) a[k] = attrs[i * 10 + k];
  float e0 = 0.f;
#pragma unroll
  for (int k = 0; k < 10; ++k) e0 += a[k] * atomic_E[k];
  atomicAdd(&e0_ws[batch[i]], e0);

  for (int c = 0; c < 64; ++c) {
    float s = 0.f;
#pragma unroll
    for (int k = 0; k < 10; ++k) s += a[k] * emb_W[k * 64 + c];
    size_t base = (size_t)i * 1024 + (size_t)c * 16;
    node_feats[base] = s;
#pragma unroll
    for (int l = 1; l < 16; ++l) node_feats[base + l] = 0.f;
  }

  float o[16];
  float r = sph16(pos[3 * i], pos[3 * i + 1], pos[3 * i + 2], o);
  float amp = amp_ws[0];
  float inv = 1.0f / (1.0f + r);
  float p1 = inv, p2 = inv * inv, p3 = p2 * inv, p4 = p3 * inv;
  const float pw[16] = {p1, p2, p2, p2, p3, p3, p3, p3, p3, p4, p4, p4, p4, p4, p4, p4};
  float t[16];
#pragma unroll
  for (int m = 0; m < 16; ++m) t[m] = tanhf(amp * o[m] * pw[m]);
#pragma unroll
  for (int l = 0; l < 16; ++l) {
    float s = 0.f;
#pragma unroll
    for (int m = 0; m < 16; ++m) s += t[m] * field_up_W[m * 16 + l];
    mpf[(size_t)i * 16 + l] = s;
  }
}

__global__ void edge_geom_kernel(const float* __restrict__ pos,
                                 const float* __restrict__ shifts,
                                 const int* __restrict__ edge_index,
                                 float* __restrict__ sh,
                                 float* __restrict__ ef, int E) {
  int e = blockIdx.x * blockDim.x + threadIdx.x;
  if (e >= E) return;
  int s = edge_index[e];
  int d = edge_index[E + e];
  float vx = pos[3 * d] - pos[3 * s] + shifts[3 * e];
  float vy = pos[3 * d + 1] - pos[3 * s + 1] + shifts[3 * e + 1];
  float vz = pos[3 * d + 2] - pos[3 * s + 2] + shifts[3 * e + 2];
  float o[16];
  float r = sph16(vx, vy, vz, o);
#pragma unroll
  for (int l = 0; l < 16; ++l) sh[(size_t)e * 16 + l] = o[l];
  float u = r * (1.0f / R_MAX);
  float u2 = u * u, u4 = u2 * u2, u5 = u4 * u, u6 = u5 * u, u7 = u6 * u;
  float env = (u < 1.0f) ? (1.0f - 21.0f * u5 + 35.0f * u6 - 15.0f * u7) : 0.0f;
  float pref = sqrtf(2.0f / R_MAX) * __builtin_amdgcn_rcpf(r + 1e-9f) * env;
  const float k = 3.14159265358979323846f / R_MAX;
#pragma unroll
  for (int nb = 1; nb <= 8; ++nb)
    ef[(size_t)e * 8 + nb - 1] = pref * __sinf((float)nb * k * r);
}

__global__ void energies_init_kernel(const float* __restrict__ e0, float* __restrict__ out, int G) {
  int i = blockIdx.x * blockDim.x + threadIdx.x;
  if (i < G * 3) out[i] = e0[i / 3];
}

// ---------------------------------------------------------------------------
// Fused edge MLP (WMMA f16) + message + scatter. 4 waves x 16 edges.
// ---------------------------------------------------------------------------
#define WPB 4
#define EPB 64

__global__ __launch_bounds__(128)
void edge_mlp_message_kernel(const float* __restrict__ ef,
                             const float* __restrict__ sh,
                             const int* __restrict__ edge_index,
                             const float* __restrict__ node_feats,
                             const float* __restrict__ W1,
                             const float* __restrict__ b1,
                             const _Float16* __restrict__ W2L,
                             const float* __restrict__ b2,
                             const _Float16* __restrict__ W3L,
                             float* __restrict__ agg,
                             int E) {
  __shared__ __align__(16) float    sW1[8 * 64];
  __shared__ __align__(16) float    sb1[64], sb2[64];
  __shared__ __align__(16) _Float16 sW2[64 * 64];
  __shared__ __align__(16) _Float16 sW3[2][64 * 64];   // double-buffered chunk
  __shared__ __align__(16) _Float16 sH[WPB][16 * 64];
  __shared__ __align__(16) float    sSrc[WPB][16 * 64];
  __shared__ __align__(16) float    sSh[WPB][16 * 16];
  __shared__ __align__(16) float    sEf[WPB][16 * 8];
  __shared__ int sRcv[WPB][16];

  const int tid = threadIdx.x;
  const int lane = tid & 31;
  const int wave = tid >> 5;
  const int ebase = blockIdx.x * EPB + wave * 16;

  for (int i = tid; i < 8 * 64; i += 128) sW1[i] = W1[i];
  for (int i = tid; i < 64; i += 128) { sb1[i] = b1[i]; sb2[i] = b2[i]; }
  for (int i = tid; i < 512; i += 128)
    ((uint4*)sW2)[i] = ((const uint4*)W2L)[i];

  {
    int i = lane;
    int m = i >> 1, part = i & 1;
    int e = ebase + m; e = e < E ? e : E - 1;
    ((float4*)sEf[wave])[i] = *(const float4*)&ef[(size_t)e * 8 + part * 4];
  }
#pragma unroll
  for (int j = 0; j < 2; ++j) {
    int i = lane + j * 32;
    int m = i >> 2, part = i & 3;
    int e = ebase + m; e = e < E ? e : E - 1;
    ((float4*)sSh[wave])[i] = *(const float4*)&sh[(size_t)e * 16 + part * 4];
  }
  if (lane < 16) {
    int e = ebase + lane; e = e < E ? e : E - 1;
    sRcv[wave][lane] = edge_index[(size_t)E + e];
  }
  for (int i = lane; i < 16 * 64; i += 32) {
    int m = i >> 6, c = i & 63;
    int e = ebase + m; e = e < E ? e : E - 1;
    int s = edge_index[e];
    sSrc[wave][i] = node_feats[(size_t)s * 1024 + (size_t)c * 16];
  }
  // prefetch chunk 0 of W3 while h1/h2 compute
  stage_w3_chunk(W3L, sW3[0], tid);
  __syncthreads();

  // h1 = silu(ef @ W1 + b1)
  {
    int m = lane >> 1;
    int cb = (lane & 1) * 32;
    float efr[8];
#pragma unroll
    for (int k = 0; k < 8; ++k) efr[k] = sEf[wave][m * 8 + k];
#pragma unroll
    for (int g = 0; g < 4; ++g) {
      v8h pack;
#pragma unroll
      for (int j = 0; j < 8; ++j) {
        int c = cb + g * 8 + j;
        float acc = sb1[c];
#pragma unroll
        for (int k = 0; k < 8; ++k) acc += efr[k] * sW1[k * 64 + c];
        pack[j] = (_Float16)fast_silu(acc);
      }
      *(v8h*)&sH[wave][m * 64 + cb + g * 8] = pack;
    }
  }
  __syncthreads();

  // h2 = silu(h1 @ W2 + b2)
  v16h a0 = load_frag_A(&sH[wave][0], 64, 0, lane);
  v16h a1 = load_frag_A(&sH[wave][0], 64, 32, lane);
  v8f acc2[4];
#pragma unroll
  for (int t = 0; t < 4; ++t) {
    v8f c = {};
    c = wmma_f16(a0, load_frag_Bsw(sW2, t * 2 + 0, lane), c);
    c = wmma_f16(a1, load_frag_Bsw(sW2, t * 2 + 1, lane), c);
    acc2[t] = c;
  }
  __syncthreads();

  {
    int n = lane & 15;
    int mh = (lane >> 4) * 8;
#pragma unroll
    for (int t = 0; t < 4; ++t) {
      int d = t * 16 + n;
#pragma unroll
      for (int r = 0; r < 8; ++r) {
        int m = mh + r;
        sH[wave][m * 64 + d] = (_Float16)fast_silu(acc2[t][r] + sb2[d]);
      }
    }
  }
  wait_w3();       // chunk 0 resident
  __syncthreads();

  // tpw = h2 @ W3 fused with message scatter; double-buffered chunks.
  v16h h0 = load_frag_A(&sH[wave][0], 64, 0, lane);
  v16h h1f = load_frag_A(&sH[wave][0], 64, 32, lane);
  const float inv_avg = 1.0f / AVG_NEIGH;

  for (int chunk = 0; chunk < 16; ++chunk) {
    int cur = chunk & 1;
    if (chunk < 15)
      stage_w3_chunk(W3L + (size_t)(chunk + 1) * 4096, sW3[cur ^ 1], tid);
#pragma unroll
    for (int t = 0; t < 4; ++t) {
      v8f c = {};
      c = wmma_f16(h0, load_frag_Bsw(sW3[cur], t * 2 + 0, lane), c);
      c = wmma_f16(h1f, load_frag_Bsw(sW3[cur], t * 2 + 1, lane), c);
      int n = lane & 15;
      int mh = (lane >> 4) * 8;
      int ch = chunk * 4 + t;
#pragma unroll
      for (int r = 0; r < 8; ++r) {
        int m = mh + r;
        if (ebase + m < E) {
          float val = c[r] * sSrc[wave][m * 64 + ch] * sSh[wave][m * 16 + n] * inv_avg;
          atomicAdd(&agg[(size_t)sRcv[wave][m] * 1024 + (size_t)ch * 16 + n], val);
        }
      }
    }
    wait_w3();
    __syncthreads();
  }
}

// ---------------------------------------------------------------------------
// Per-node update.  Block = 128 threads = 4 waves; wave w owns N-tile w.
// Three M=16(l) x K=64(c) x N=64(d) GEMMs in full f32 via v_wmma_f32_16x16x4_f32.
// ---------------------------------------------------------------------------
#if defined(HAVE_WMMA_F32)
// swizzled B index for f32 WMMA: frag for (t, kk) holds lane-major pairs.
__device__ __forceinline__ int swz_b32(int c, int d) {
  int kk = c >> 2, r2 = c & 3, half = r2 >> 1, w = r2 & 1;
  int t = d >> 4, n = d & 15, lane = half * 16 + n;
  return ((t * 16 + kk) * 32 + lane) * 2 + w;
}
__device__ __forceinline__ v2f load_fragB32(const float* ld, int t, int kk, int lane) {
  return *(const v2f*)&ld[((t * 16 + kk) * 32 + lane) * 2];
}
// A fragment: pair from transposed row-major [16][64] array.
__device__ __forceinline__ v2f load_fragA32(const float* ld, int kk, int lane) {
  int m = lane & 15, half = lane >> 4;
  return *(const v2f*)&ld[m * 64 + kk * 4 + 2 * half];
}
#endif

__global__ __launch_bounds__(128)
void node_update_kernel(const float* __restrict__ agg,
                        const float* __restrict__ attrs,
                        const float* __restrict__ mpf,
                        const float* __restrict__ mix_W,
                        const float* __restrict__ sc_W,
                        const float* __restrict__ field_W,
                        const float* __restrict__ prod_W,
                        const float* __restrict__ readout_W,
                        const int* __restrict__ batch,
                        float* __restrict__ node_feats,
                        float* __restrict__ out) {
  __shared__ __align__(16) float sAggT[16 * 64];   // [l][c]; reused as sOut [l][d]
  __shared__ __align__(16) float sFeatT[16 * 64];  // [l][c]
  __shared__ __align__(16) float sNfT[16 * 64];    // [l][d]
  __shared__ __align__(16) float sMixS[64 * 64];
  __shared__ __align__(16) float sProdS[64 * 64];
  __shared__ __align__(16) float sWnS[64 * 64];
  __shared__ float sFw[64], sMpf[16], sAttr[16];

  const int nid = blockIdx.x;
  const int tid = threadIdx.x;
  const size_t nbase = (size_t)nid * 1024;
#if defined(HAVE_WMMA_F32)
  const int lane = tid & 31;
  const int wv = tid >> 5;   // N-tile owned by this wave
#endif

  for (int i = tid; i < 1024; i += 128) {
    int c = i >> 4, l = i & 15;       // global order: c-major, l-minor
    sAggT[l * 64 + c] = agg[nbase + i];
    sFeatT[l * 64 + c] = node_feats[nbase + i];
  }
  for (int i = tid; i < 4096; i += 128) {
    int c = i >> 6, d = i & 63;
#if defined(HAVE_WMMA_F32)
    sMixS[swz_b32(c, d)] = mix_W[i];
    sProdS[swz_b32(c, d)] = prod_W[i];
#else
    (void)c; (void)d;
    sMixS[i] = mix_W[i];
    sProdS[i] = prod_W[i];
#endif
  }
  if (tid < 16) sMpf[tid] = mpf[(size_t)nid * 16 + tid];
  if (tid < 10) sAttr[tid] = attrs[(size_t)nid * 10 + tid];
  __syncthreads();

  // W_n[c][d] = sum_k attrs[k] * sc_W[k][c][d]  (stored swizzled for WMMA)
  for (int i = tid; i < 4096; i += 128) {
    float s = 0.f;
#pragma unroll
    for (int k = 0; k < 10; ++k) s += sAttr[k] * sc_W[k * 4096 + i];
#if defined(HAVE_WMMA_F32)
    sWnS[swz_b32(i >> 6, i & 63)] = s;
#else
    sWnS[i] = s;
#endif
  }
  // fW[d] = sum_c feats[c][0] * field_W[c][d]   (feats[c][0] = sFeatT[0][c])
  if (tid < 64) {
    float s = 0.f;
    for (int c = 0; c < 64; ++c) s += sFeatT[c] * field_W[c * 64 + tid];
    sFw[tid] = s;
  }
  __syncthreads();

#if defined(HAVE_WMMA_F32)
  // nfT[l][d] = sum_c aggT[l][c]*mix[c][d] + fW[d]*mpf[l]
  {
    v8f acc = {};
    for (int kk = 0; kk < 16; ++kk)
      acc = wmma_f32(load_fragA32(sAggT, kk, lane), load_fragB32(sMixS, wv, kk, lane), acc);
    int n = lane & 15, half = lane >> 4;
    int d = wv * 16 + n;
#pragma unroll
    for (int r = 0; r < 8; ++r) {
      int l = r + 8 * half;
      sNfT[l * 64 + d] = acc[r] + sFw[d] * sMpf[l];
    }
  }
  __syncthreads();

  // outT[l][d] = sum_c (nfT[l][c]*nfT[0][c]) * prod[c][d] + sum_c featT[l][c]*Wn[c][d]
  {
    v8f acc = {};
    for (int kk = 0; kk < 16; ++kk) {
      int m = lane & 15, half = lane >> 4;
      int kb = kk * 4 + 2 * half;
      v2f a = *(const v2f*)&sNfT[m * 64 + kb];
      v2f z = *(const v2f*)&sNfT[kb];          // row l=0
      v2f a2 = a * z;
      acc = wmma_f32(a2, load_fragB32(sProdS, wv, kk, lane), acc);
    }
    for (int kk = 0; kk < 16; ++kk)
      acc = wmma_f32(load_fragA32(sFeatT, kk, lane), load_fragB32(sWnS, wv, kk, lane), acc);
    __syncthreads();          // sAggT reads done; reuse as sOut
    int n = lane & 15, half = lane >> 4;
    int d = wv * 16 + n;
#pragma unroll
    for (int r = 0; r < 8; ++r) {
      int l = r + 8 * half;
      sAggT[l * 64 + d] = acc[r];
    }
  }
  __syncthreads();
#else
  for (int i = tid; i < 1024; i += 128) {
    int l = i >> 6, d = i & 63;
    float s = sFw[d] * sMpf[l];
    for (int c = 0; c < 64; ++c) s += sAggT[l * 64 + c] * sMixS[c * 64 + d];
    sNfT[l * 64 + d] = s;
  }
  __syncthreads();
  for (int i = tid; i < 1024; i += 128) {
    int l = i >> 6, d = i & 63;
    float s = 0.f;
    for (int c = 0; c < 64; ++c) {
      s += (sNfT[l * 64 + c] * sNfT[c]) * sProdS[c * 64 + d];
      s += sFeatT[l * 64 + c] * sWnS[c * 64 + d];
    }
    sNfT[l * 64 + d] = s;    // careful ordering not needed in fallback path:
  }
  __syncthreads();
  for (int i = tid; i < 1024; i += 128) sAggT[i] = sNfT[i];  // unify output buffer
  __syncthreads();
#endif

  // write back (coalesced) + readout
  for (int i = tid; i < 1024; i += 128) {
    int c = i >> 4, l = i & 15;
    node_feats[nbase + i] = sAggT[l * 64 + c];
  }
  __syncthreads();
  if (tid < 3) {
    float s = 0.f;
    for (int d = 0; d < 64; ++d) s += sAggT[d] * readout_W[d * 3 + tid];
    atomicAdd(&out[(size_t)batch[nid] * 3 + tid], s);
  }
}

// ---------------------------------------------------------------------------
extern "C" void kernel_launch(void* const* d_in, const int* in_sizes, int n_in,
                              void* d_out, int out_size, void* d_ws, size_t ws_size,
                              hipStream_t stream) {
  const float* positions  = (const float*)d_in[0];
  const float* node_attrs = (const float*)d_in[1];
  const float* shifts     = (const float*)d_in[2];
  const float* mm_pos     = (const float*)d_in[3];
  const float* mm_q       = (const float*)d_in[4];
  const int*   edge_index = (const int*)d_in[5];
  const int*   batch      = (const int*)d_in[6];
  const float* emb_W      = (const float*)d_in[8];
  const float* atomic_E   = (const float*)d_in[9];
  const float* rad_W1     = (const float*)d_in[10];
  const float* rad_b1     = (const float*)d_in[11];
  const float* rad_W2     = (const float*)d_in[12];
  const float* rad_b2     = (const float*)d_in[13];
  const float* rad_W3     = (const float*)d_in[14];
  const float* mix_W      = (const float*)d_in[15];
  const float* sc_W       = (const float*)d_in[16];
  const float* field_W    = (const float*)d_in[17];
  const float* prod_W     = (const float*)d_in[18];
  const float* readout_W  = (const float*)d_in[19];
  const float* field_up_W = (const float*)d_in[20];

  const int N = in_sizes[0] / 3;
  const int E = in_sizes[5] / 2;
  const int M = in_sizes[4];
  const int G = out_size / 3;
  float* out = (float*)d_out;

  float* f = (float*)d_ws;
  size_t off = 0;
  float* sh_ws    = f + off; off += (size_t)E * 16;
  float* ef_ws    = f + off; off += (size_t)E * 8;
  float* feats_ws = f + off; off += (size_t)N * 1024;
  float* agg_ws   = f + off; off += (size_t)N * 1024;
  float* mpf_ws   = f + off; off += (size_t)N * 16;
  float* e0_ws    = f + off; off += 64;
  float* amp_ws   = f + off; off += 16;
  _Float16* W2h   = (_Float16*)(f + off);
  _Float16* W3h   = W2h + 2 * 4096;

  swz_W2_kernel<<<(2 * 4096 + 255) / 256, 256, 0, stream>>>(rad_W2, W2h);
  swz_W3_kernel<<<(2 * 65536 + 255) / 256, 256, 0, stream>>>(rad_W3, W3h);

  zero_kernel<<<1, 64, 0, stream>>>(e0_ws, (size_t)G);
  mm_moments_kernel<<<1, 256, 0, stream>>>(mm_pos, mm_q, amp_ws, M);

  node_init_kernel<<<(N + 255) / 256, 256, 0, stream>>>(
      positions, node_attrs, emb_W, atomic_E, field_up_W, amp_ws, batch,
      feats_ws, mpf_ws, e0_ws, N);

  edge_geom_kernel<<<(E + 255) / 256, 256, 0, stream>>>(
      positions, shifts, edge_index, sh_ws, ef_ws, E);

  energies_init_kernel<<<1, 64, 0, stream>>>(e0_ws, out, G);

  for (int layer = 0; layer < 2; ++layer) {
    zero_kernel<<<(int)(((size_t)N * 1024 + 255) / 256), 256, 0, stream>>>(
        agg_ws, (size_t)N * 1024);

    edge_mlp_message_kernel<<<(E + EPB - 1) / EPB, 128, 0, stream>>>(
        ef_ws, sh_ws, edge_index, feats_ws,
        rad_W1 + layer * 8 * 64, rad_b1 + layer * 64,
        W2h + layer * 4096, rad_b2 + layer * 64,
        W3h + (size_t)layer * 65536,
        agg_ws, E);

    node_update_kernel<<<N, 128, 0, stream>>>(
        agg_ws, node_attrs, mpf_ws,
        mix_W + layer * 4096, sc_W + layer * 40960,
        field_W + layer * 4096, prod_W + layer * 4096,
        readout_W + layer * 192, batch,
        feats_ws, out);
  }
}
